// GATModel3_13804024889637
// MI455X (gfx1250) — matrix operand order repaired and checked by reading, not compiled, not run on hardware
//
#include <hip/hip_runtime.h>
#include <hip/hip_bf16.h>
#include <math.h>

#define N_NODES 100000
#define N_EDGES 1600000

typedef __attribute__((ext_vector_type(16))) __bf16 v16bf;
typedef __attribute__((ext_vector_type(8)))  float  v8f;

// ---------------------------------------------------------------------------
// GEMM: C[M,Nc] = A[M,K] * B[K,Nc], f32 in/out, bf16 WMMA, f32 accumulate.
// One wave computes a 32x64 macro-tile (2 M-tiles x 4 N-tiles = 8 WMMA accs),
// reusing A fragments across 4 N-tiles and B fragments across 2 M-tiles.
// M%32==0 (100000=32*3125), K%32==0, Nc%64==0 guaranteed by the model dims.
// Fragment layouts per CDNA5 ISA 7.12.2 (wave32).
// ---------------------------------------------------------------------------
__global__ __launch_bounds__(256)
void gemm_bf16_wmma(const float* __restrict__ A, const float* __restrict__ B,
                    float* __restrict__ C, int M, int K, int Nc) {
    int gwave  = (int)((blockIdx.x * (size_t)blockDim.x + threadIdx.x) >> 5);
    int lane   = threadIdx.x & 31;
    int groupsN = Nc >> 6;                 // 64-wide N groups
    int groupsM = M  >> 5;                 // 32-wide M groups
    int gM = gwave / groupsN;
    int gN = gwave % groupsN;
    if (gM >= groupsM) return;

    int half = lane >> 4;                  // 0: lanes 0-15, 1: lanes 16-31
    int lid  = lane & 15;
    int m0 = gM << 5;
    int n0 = gN << 6;

    v8f acc[2][4] = {};
    for (int kk = 0; kk < K; kk += 32) {
        v16bf a[2], b[4];
        // A fragments: lane holds row (m0 + mi*16 + lid); K pairs per ISA table
#pragma unroll
        for (int mi = 0; mi < 2; ++mi) {
            const float* Arow = A + (size_t)(m0 + mi * 16 + lid) * K + kk;
#pragma unroll
            for (int i = 0; i < 8; ++i) {
                int kb = ((i & 4) ? 16 : 0) + (half ? 8 : 0) + ((i & 3) << 1);
                a[mi][2 * i]     = (__bf16)Arow[kb];
                a[mi][2 * i + 1] = (__bf16)Arow[kb + 1];
            }
        }
        // B fragments: lane holds col (n0 + ni*16 + lid); K pairs per ISA table (same as A)
        int koff = half ? 16 : 0;
#pragma unroll
        for (int ni = 0; ni < 4; ++ni) {
            const float* Bcol = B + (size_t)kk * Nc + n0 + ni * 16 + lid;
#pragma unroll
            for (int i = 0; i < 8; ++i) {
                int kb = ((i & 4) ? 16 : 0) + (half ? 8 : 0) + ((i & 3) << 1);
                b[ni][2 * i]     = (__bf16)Bcol[(size_t)kb * Nc];
                b[ni][2 * i + 1] = (__bf16)Bcol[(size_t)(kb + 1) * Nc];
            }
        }
#pragma unroll
        for (int mi = 0; mi < 2; ++mi)
#pragma unroll
            for (int ni = 0; ni < 4; ++ni)
                acc[mi][ni] = __builtin_amdgcn_wmma_f32_16x16x32_bf16(
                    false, a[mi], false, b[ni], (short)0, acc[mi][ni], false, false);
    }
    // C: VGPR i -> row (half*8 + i), lane -> col
#pragma unroll
    for (int mi = 0; mi < 2; ++mi) {
#pragma unroll
        for (int ni = 0; ni < 4; ++ni) {
            float* Crow = C + (size_t)(m0 + mi * 16 + half * 8) * Nc + n0 + ni * 16 + lid;
#pragma unroll
            for (int i = 0; i < 8; ++i) Crow[(size_t)i * Nc] = acc[mi][ni][i];
        }
    }
}

// ---------------------------------------------------------------------------
// el/er attention scores: el[n,h] = sum_d z[n,h,d]*al[h,d]
// ---------------------------------------------------------------------------
__global__ void scores_k(const float* __restrict__ z, const float* __restrict__ al,
                         const float* __restrict__ ar, float* __restrict__ el,
                         float* __restrict__ er, int H, int D) {
    int i = blockIdx.x * blockDim.x + threadIdx.x;   // over N*H
    if (i >= N_NODES * H) return;
    int h = i % H;
    const float* zp = z + (size_t)i * D;
    float sl = 0.f, sr = 0.f;
    for (int d = 0; d < D; ++d) {
        float v = zp[d];
        sl += v * al[h * D + d];
        sr += v * ar[h * D + d];
    }
    el[i] = sl;
    er[i] = sr;
}

__global__ void init_md_k(float* __restrict__ m, float* __restrict__ dn, int n) {
    int i = blockIdx.x * blockDim.x + threadIdx.x;
    if (i < n) { m[i] = -3.0e38f; dn[i] = 0.f; }
}

__global__ void zero_k(float* __restrict__ p, int n) {
    int i = blockIdx.x * blockDim.x + threadIdx.x;
    if (i < n) p[i] = 0.f;
}

__device__ inline void atomicMaxF(float* addr, float val) {
    int* p = (int*)addr;
    int cur = __float_as_int(*addr);
    while (__int_as_float(cur) < val) {
        int prev = atomicCAS(p, cur, __float_as_int(val));
        if (prev == cur) break;
        cur = prev;
    }
}

// pass A: e = leaky_relu(el[src]+er[dst]); store e; segment max over dst
__global__ void edge_max_k(const int* __restrict__ src, const int* __restrict__ dst,
                           const float* __restrict__ el, const float* __restrict__ er,
                           float* __restrict__ ebuf, float* __restrict__ mmax, int H) {
    int i = blockIdx.x * blockDim.x + threadIdx.x;   // over E*H
    if (i >= N_EDGES * H) return;
    int e = i / H, h = i - e * H;
    float v = el[src[e] * H + h] + er[dst[e] * H + h];
    v = (v > 0.f) ? v : 0.2f * v;
    ebuf[i] = v;
    atomicMaxF(&mmax[dst[e] * H + h], v);
}

// pass B: ex = exp(e - m[dst]); segment sum denom
__global__ void edge_exp_k(const int* __restrict__ dst, float* __restrict__ ebuf,
                           const float* __restrict__ mmax, float* __restrict__ denom, int H) {
    int i = blockIdx.x * blockDim.x + threadIdx.x;
    if (i >= N_EDGES * H) return;
    int e = i / H, h = i - e * H;
    int d = dst[e];
    float ex = __expf(ebuf[i] - mmax[d * H + h]);
    ebuf[i] = ex;
    atomicAdd(&denom[d * H + h], ex);
}

// pass C: agg[dst] += z[src] * (ex / denom[dst])
__global__ void edge_agg_k(const int* __restrict__ src, const int* __restrict__ dst,
                           const float* __restrict__ ebuf, const float* __restrict__ denom,
                           const float* __restrict__ z, float* __restrict__ agg,
                           int H, int D) {
    int i = blockIdx.x * blockDim.x + threadIdx.x;   // over E*H
    if (i >= N_EDGES * H) return;
    int e = i / H, h = i - e * H;
    int s = src[e], d = dst[e];
    float alpha = ebuf[i] / denom[d * H + h];
    const float* zp = z + ((size_t)s * H + h) * D;
    float* op = agg + ((size_t)d * H + h) * D;
    for (int j = 0; j < D; ++j) atomicAdd(&op[j], zp[j] * alpha);
}

__global__ void relu_bias_k(float* __restrict__ x, const float* __restrict__ b, int HD) {
    int i = blockIdx.x * blockDim.x + threadIdx.x;   // over N*HD
    if (i >= N_NODES * HD) return;
    float v = x[i] + b[i % HD];
    x[i] = (v > 0.f) ? v : 0.f;
}

__global__ void head_k(const float* __restrict__ x, const float* __restrict__ Wp,
                       const float* __restrict__ bp, float* __restrict__ out) {
    int n = blockIdx.x * blockDim.x + threadIdx.x;
    if (n >= N_NODES) return;
    const float* xp = x + (size_t)n * 128;
    float s = 0.f;
    for (int j = 0; j < 128; ++j) s += xp[j] * Wp[j];
    s += bp[0];
    out[n] = 1.f / (1.f + __expf(-s));
}

// ---------------------------------------------------------------------------
static inline int cdiv(long long a, long long b) { return (int)((a + b - 1) / b); }

extern "C" void kernel_launch(void* const* d_in, const int* in_sizes, int n_in,
                              void* d_out, int out_size, void* d_ws, size_t ws_size,
                              hipStream_t stream) {
    const float* feats = (const float*)d_in[0];
    const int*   src   = (const int*)d_in[1];
    const int*   dst   = (const int*)d_in[2];
    // d_in[3] edge_types: unused by reference forward
    const float* W1  = (const float*)d_in[4];
    const float* al1 = (const float*)d_in[5];
    const float* ar1 = (const float*)d_in[6];
    const float* b1  = (const float*)d_in[7];
    const float* W2  = (const float*)d_in[8];
    const float* al2 = (const float*)d_in[9];
    const float* ar2 = (const float*)d_in[10];
    const float* b2  = (const float*)d_in[11];
    const float* W3  = (const float*)d_in[12];
    const float* al3 = (const float*)d_in[13];
    const float* ar3 = (const float*)d_in[14];
    const float* b3  = (const float*)d_in[15];
    const float* Wp  = (const float*)d_in[16];
    const float* bp  = (const float*)d_in[17];
    float* out = (float*)d_out;

    // workspace layout (floats)
    float* ws = (float*)d_ws;
    const size_t NA = (size_t)N_NODES * 256;   // bufA
    const size_t NB = (size_t)N_NODES * 256;   // bufB
    const size_t NC = (size_t)N_NODES * 128;   // bufC
    const size_t NE = (size_t)N_EDGES * 8;     // edge scratch (max H=8)
    const size_t NH = (size_t)N_NODES * 8;     // per-node per-head
    float* bufA = ws;
    float* bufB = bufA + NA;
    float* bufC = bufB + NB;
    float* ex   = bufC + NC;
    float* el   = ex + NE;
    float* er   = el + NH;
    float* mm   = er + NH;
    float* dn   = mm + NH;

    const int T = 256;
    auto gemm = [&](const float* A, const float* B, float* C, int K, int Ncol) {
        long long waves = (long long)(N_NODES / 32) * (Ncol / 64);
        gemm_bf16_wmma<<<cdiv(waves * 32, T), T, 0, stream>>>(A, B, C, N_NODES, K, Ncol);
    };
    auto layer = [&](const float* X, const float* W, const float* al, const float* ar,
                     const float* b, float* Z, float* AGG, int K, int H, int D) {
        int HD = H * D;
        gemm(X, W, Z, K, HD);
        scores_k<<<cdiv((long long)N_NODES * H, T), T, 0, stream>>>(Z, al, ar, el, er, H, D);
        init_md_k<<<cdiv((long long)N_NODES * H, T), T, 0, stream>>>(mm, dn, N_NODES * H);
        edge_max_k<<<cdiv((long long)N_EDGES * H, T), T, 0, stream>>>(src, dst, el, er, ex, mm, H);
        edge_exp_k<<<cdiv((long long)N_EDGES * H, T), T, 0, stream>>>(dst, ex, mm, dn, H);
        zero_k<<<cdiv((long long)N_NODES * HD, T), T, 0, stream>>>(AGG, N_NODES * HD);
        edge_agg_k<<<cdiv((long long)N_EDGES * H, T), T, 0, stream>>>(src, dst, ex, dn, Z, AGG, H, D);
        relu_bias_k<<<cdiv((long long)N_NODES * HD, T), T, 0, stream>>>(AGG, b, HD);
    };

    // Layer 1: 256 -> 8x32 ; X=features, Z=bufA, AGG=bufB
    layer(feats, W1, al1, ar1, b1, bufA, bufB, 256, 8, 32);
    // Layer 2: 256 -> 4x32 ; X=bufB, Z=bufA, AGG=bufC
    layer(bufB, W2, al2, ar2, b2, bufA, bufC, 256, 4, 32);
    // Layer 3: 128 -> 1x128 ; X=bufC, Z=bufA, AGG=bufB
    layer(bufC, W3, al3, ar3, b3, bufA, bufB, 128, 1, 128);
    // Prediction head
    head_k<<<cdiv(N_NODES, T), T, 0, stream>>>(bufB, Wp, bp, out);
}
